// lstm_wrapper_15539191677047
// MI455X (gfx1250) — matrix operand order repaired and checked by reading, not compiled, not run on hardware
//
#include <hip/hip_runtime.h>
#include <hip/hip_bf16.h>

#define HID   1024
#define INSZ  64
#define BATCH 128
#define SEQ   256
#define TGT   96

typedef __attribute__((ext_vector_type(16))) _Float16 v16h;
typedef __attribute__((ext_vector_type(8)))  float    v8f;

union Frag16 {
    v16h h;
    uint4 u[2];
};

// ---------------------------------------------------------------------------
// Fragment loaders matching the CDNA5 WMMA VGPR layouts (CDNA5 ISA)
// A 16x32 f16 (MxK): lane m = lane&15; lanes0-15 kbase=0, lanes16-31 kbase=8;
//   V0..V3 hold K = kbase+0..7, V4..V7 hold K = 16+kbase+0..7.
// B 32x16 f16 (KxN): lane n = lane&15; lanes0-15 kbase=0, lanes16-31 kbase=8;
//   K = kbase+0..7 and 16+kbase+0..7 -> two b128 loads from row-major W[N][K].
// Loaders take a per-lane row base pointer so the k offset lowers to an
// immediate IOFFSET in the fully-unrolled loops (no per-tile address ALU).
// ---------------------------------------------------------------------------
__device__ __forceinline__ v16h load_a_at(const _Float16* __restrict__ laneRow, int k0) {
    // laneRow = base + (row0 + lane%16)*ldk + (lane/16)*8
    Frag16 f;
    f.u[0] = *reinterpret_cast<const uint4*>(laneRow + k0);
    f.u[1] = *reinterpret_cast<const uint4*>(laneRow + k0 + 16);
    return f.h;
}

__device__ __forceinline__ v16h load_b_at(const _Float16* __restrict__ laneRow, int k0) {
    // laneRow = base + (n0 + lane%16)*ldk + (lane/16)*8
    Frag16 f;
    f.u[0] = *reinterpret_cast<const uint4*>(laneRow + k0);
    f.u[1] = *reinterpret_cast<const uint4*>(laneRow + k0 + 16);
    return f.h;
}

__device__ __forceinline__ v8f wmma_f16(v16h a, v16h b, v8f c) {
    return __builtin_amdgcn_wmma_f32_16x16x32_f16(false, a, false, b,
                                                  (short)0, c, false, false);
}

__device__ __forceinline__ float sigm(float x) {
    return 1.0f / (1.0f + __expf(-x));
}

__device__ __forceinline__ float fast_tanh(float x) {
#if __has_builtin(__builtin_amdgcn_tanhf)
    return __builtin_amdgcn_tanhf(x);   // gfx1250 v_tanh_f32
#else
    return tanhf(x);
#endif
}

// ---------------------------------------------------------------------------
// Fused LSTM step: z = h@Whh^T + x@Wih^T + (bih+bhh); gates; c,h update.
// Grid (16, 8), block 128 (4 waves). One wave -> one 16-col slice of H, all
// four gates as 4 accumulators sharing one A fragment per K-step.
// Fully-unrolled software pipeline: tile kt+32's A + 4xB fragments load into
// fresh (SSA) registers before tile kt's WMMAs, giving load/WMMA overlap with
// no register copies, no hazard NOPs, and immediate-offset addressing.
// ---------------------------------------------------------------------------
__global__ __launch_bounds__(128)
void lstm_step_kernel(const _Float16* __restrict__ x16,   // [128][64] fp16
                      const _Float16* __restrict__ Wih,   // [4096][64] fp16
                      const _Float16* __restrict__ Whh,   // [4096][1024] fp16
                      const float*    __restrict__ bias,  // [4096] (bih+bhh)
                      const _Float16* __restrict__ h_in,  // [128][1024] fp16
                      float*          __restrict__ c_io,  // [128][1024] f32
                      _Float16*       __restrict__ h_out) // [128][1024] fp16
{
    const int lane = threadIdx.x & 31;
    const int wave = threadIdx.x >> 5;
    const int jt   = blockIdx.x * 4 + wave;   // col tile 0..63 within H
    const int rt   = blockIdx.y;              // row tile 0..7 (batch)
    const int n0   = jt * 16;
    const int col  = n0 + (lane & 15);
    const int row0 = rt * 16;

    // Per-lane base pointers (all k offsets become immediates below)
    const int laneN = lane & 15;
    const int aOff  = (lane >> 4) << 3;   // A: +0 or +8 halves
    const int bOff  = (lane >> 4) << 3;   // B: +0 or +8 halves
    const _Float16* hRow  = h_in + (size_t)(row0 + laneN) * HID + aOff;
    const _Float16* xRow  = x16  + (size_t)(row0 + laneN) * INSZ + aOff;
    const _Float16* w0Row = Whh + (size_t)(0 * HID + n0 + laneN) * HID + bOff;
    const _Float16* w1Row = Whh + (size_t)(1 * HID + n0 + laneN) * HID + bOff;
    const _Float16* w2Row = Whh + (size_t)(2 * HID + n0 + laneN) * HID + bOff;
    const _Float16* w3Row = Whh + (size_t)(3 * HID + n0 + laneN) * HID + bOff;
    const _Float16* u0Row = Wih + (size_t)(0 * HID + n0 + laneN) * INSZ + bOff;
    const _Float16* u1Row = Wih + (size_t)(1 * HID + n0 + laneN) * INSZ + bOff;
    const _Float16* u2Row = Wih + (size_t)(2 * HID + n0 + laneN) * INSZ + bOff;
    const _Float16* u3Row = Wih + (size_t)(3 * HID + n0 + laneN) * INSZ + bOff;

    // Warm the epilogue's strided c lines while the GEMM runs
    const int rbase = row0 + ((lane >> 4) << 3);
#pragma unroll
    for (int r = 0; r < 8; ++r) {
        __builtin_prefetch(c_io + (size_t)(rbase + r) * HID + col, 0, 3);
    }

    const float bi = bias[0 * HID + col];
    const float bf = bias[1 * HID + col];
    const float bg = bias[2 * HID + col];
    const float bo = bias[3 * HID + col];
    v8f acc0 = (v8f){bi, bi, bi, bi, bi, bi, bi, bi};
    v8f acc1 = (v8f){bf, bf, bf, bf, bf, bf, bf, bf};
    v8f acc2 = (v8f){bg, bg, bg, bg, bg, bg, bg, bg};
    v8f acc3 = (v8f){bo, bo, bo, bo, bo, bo, bo, bo};

    // ---- prologue: fragments for kt = 0 ----
    v16h a  = load_a_at(hRow, 0);
    v16h b0 = load_b_at(w0Row, 0);
    v16h b1 = load_b_at(w1Row, 0);
    v16h b2 = load_b_at(w2Row, 0);
    v16h b3 = load_b_at(w3Row, 0);

    // ---- pipelined recurrent GEMM: K = 1024, fully unrolled ----
#pragma unroll
    for (int kt = 32; kt < HID; kt += 32) {
        v16h an = load_a_at(hRow, kt);
        v16h c0 = load_b_at(w0Row, kt);
        v16h c1 = load_b_at(w1Row, kt);
        v16h c2 = load_b_at(w2Row, kt);
        v16h c3 = load_b_at(w3Row, kt);
        acc0 = wmma_f16(a, b0, acc0);
        acc1 = wmma_f16(a, b1, acc1);
        acc2 = wmma_f16(a, b2, acc2);
        acc3 = wmma_f16(a, b3, acc3);
        a = an; b0 = c0; b1 = c1; b2 = c2; b3 = c3;
    }

    // ---- input GEMM tile 0 loads overlap with final recurrent WMMAs ----
    {
        v16h an = load_a_at(xRow, 0);
        v16h c0 = load_b_at(u0Row, 0);
        v16h c1 = load_b_at(u1Row, 0);
        v16h c2 = load_b_at(u2Row, 0);
        v16h c3 = load_b_at(u3Row, 0);
        acc0 = wmma_f16(a, b0, acc0);
        acc1 = wmma_f16(a, b1, acc1);
        acc2 = wmma_f16(a, b2, acc2);
        acc3 = wmma_f16(a, b3, acc3);
        a = an; b0 = c0; b1 = c1; b2 = c2; b3 = c3;
    }
    // ---- input GEMM tile 1 loads overlap with input tile 0 WMMAs ----
    {
        v16h an = load_a_at(xRow, 32);
        v16h c0 = load_b_at(u0Row, 32);
        v16h c1 = load_b_at(u1Row, 32);
        v16h c2 = load_b_at(u2Row, 32);
        v16h c3 = load_b_at(u3Row, 32);
        acc0 = wmma_f16(a, b0, acc0);
        acc1 = wmma_f16(a, b1, acc1);
        acc2 = wmma_f16(a, b2, acc2);
        acc3 = wmma_f16(a, b3, acc3);
        a = an; b0 = c0; b1 = c1; b2 = c2; b3 = c3;
    }
    acc0 = wmma_f16(a, b0, acc0);
    acc1 = wmma_f16(a, b1, acc1);
    acc2 = wmma_f16(a, b2, acc2);
    acc3 = wmma_f16(a, b3, acc3);

    // Gate nonlinearities + state update on the C/D VGPR layout:
    // VGPR r holds row rt*16 + r (lanes 0-15) / rt*16 + 8 + r (lanes 16-31).
#pragma unroll
    for (int r = 0; r < 8; ++r) {
        const int row = rbase + r;
        const size_t idx = (size_t)row * HID + col;
        const float zi = acc0[r];
        const float zf = acc1[r];
        const float zg = acc2[r];
        const float zo = acc3[r];
        const float c  = c_io[idx];
        const float cn = sigm(zf) * c + sigm(zi) * fast_tanh(zg);
        const float hn = sigm(zo) * fast_tanh(cn);
        c_io[idx]  = cn;
        h_out[idx] = (_Float16)hn;
    }
}

// ---------------------------------------------------------------------------
// Decoder output projection: y = h @ lin_W^T + lin_b  ([128,1024]x[1024,64])
// Grid 8 (row tiles), block 128 (4 waves = 4 col tiles of 16 -> N=64).
// Fully-unrolled software pipeline like the step kernel.
// ---------------------------------------------------------------------------
__global__ __launch_bounds__(128)
void dec_out_kernel(const _Float16* __restrict__ h,      // [128][1024] fp16
                    const _Float16* __restrict__ linW,   // [64][1024] fp16
                    const float*    __restrict__ lin_b,  // [64]
                    float*          __restrict__ y,      // [128][64] f32
                    _Float16*       __restrict__ x_next) // [128][64] fp16
{
    const int lane = threadIdx.x & 31;
    const int nt   = threadIdx.x >> 5;   // 0..3
    const int rt   = blockIdx.x;         // 0..7
    const int n0   = nt * 16;
    const int col  = n0 + (lane & 15);
    const int row0 = rt * 16;

    const int laneN = lane & 15;
    const _Float16* hRow = h    + (size_t)(row0 + laneN) * HID + (((lane >> 4) << 3));
    const _Float16* wRow = linW + (size_t)(n0 + laneN) * HID + (((lane >> 4) << 3));

    const float bv = lin_b[col];
    v8f acc = (v8f){bv, bv, bv, bv, bv, bv, bv, bv};

    v16h a = load_a_at(hRow, 0);
    v16h b = load_b_at(wRow, 0);
#pragma unroll
    for (int kt = 32; kt < HID; kt += 32) {
        v16h an = load_a_at(hRow, kt);
        v16h bn = load_b_at(wRow, kt);
        acc = wmma_f16(a, b, acc);
        a = an; b = bn;
    }
    acc = wmma_f16(a, b, acc);

    const int rbase = row0 + ((lane >> 4) << 3);
#pragma unroll
    for (int r = 0; r < 8; ++r) {
        const int row = rbase + r;
        const float v = acc[r];
        y[(size_t)row * INSZ + col]      = v;
        x_next[(size_t)row * INSZ + col] = (_Float16)v;
    }
}

// ---------------------------------------------------------------------------
// One-time setup kernels
// ---------------------------------------------------------------------------
__global__ void cvt_f32_f16(const float* __restrict__ s, _Float16* __restrict__ d, int n) {
    int i = blockIdx.x * blockDim.x + threadIdx.x;
    if (i < n) d[i] = (_Float16)s[i];
}

__global__ void fuse_bias(const float* __restrict__ a, const float* __restrict__ b,
                          float* __restrict__ o, int n) {
    int i = blockIdx.x * blockDim.x + threadIdx.x;
    if (i < n) o[i] = a[i] + b[i];
}

__global__ void zero_state(_Float16* __restrict__ h, float* __restrict__ c, int n) {
    int i = blockIdx.x * blockDim.x + threadIdx.x;
    if (i < n) { h[i] = (_Float16)0.0f; c[i] = 0.0f; }
}

// ---------------------------------------------------------------------------
extern "C" void kernel_launch(void* const* d_in, const int* in_sizes, int n_in,
                              void* d_out, int out_size, void* d_ws, size_t ws_size,
                              hipStream_t stream) {
    (void)in_sizes; (void)n_in; (void)out_size; (void)ws_size;

    const float* input_batch = (const float*)d_in[0];
    // d_in[1] target_batch unused (decoder is recursive, no teacher forcing)
    const float* enc_Wih = (const float*)d_in[2];
    const float* enc_Whh = (const float*)d_in[3];
    const float* enc_bih = (const float*)d_in[4];
    const float* enc_bhh = (const float*)d_in[5];
    const float* dec_Wih = (const float*)d_in[6];
    const float* dec_Whh = (const float*)d_in[7];
    const float* dec_bih = (const float*)d_in[8];
    const float* dec_bhh = (const float*)d_in[9];
    const float* lin_W   = (const float*)d_in[10];
    const float* lin_b   = (const float*)d_in[11];
    float* out = (float*)d_out;

    // Carve workspace (hipMalloc base is 256B aligned)
    char* ws = (char*)d_ws;
    size_t off = 0;
    auto carve = [&](size_t bytes) -> char* {
        char* p = ws + off;
        off += (bytes + 255) & ~(size_t)255;
        return p;
    };

    _Float16* Whh_e16 = (_Float16*)carve((size_t)4 * HID * HID * 2);
    _Float16* Whh_d16 = (_Float16*)carve((size_t)4 * HID * HID * 2);
    _Float16* Wih_e16 = (_Float16*)carve((size_t)4 * HID * INSZ * 2);
    _Float16* Wih_d16 = (_Float16*)carve((size_t)4 * HID * INSZ * 2);
    _Float16* linW16  = (_Float16*)carve((size_t)INSZ * HID * 2);
    float*    bias_e  = (float*)carve((size_t)4 * HID * 4);
    float*    bias_d  = (float*)carve((size_t)4 * HID * 4);
    _Float16* x16     = (_Float16*)carve((size_t)SEQ * BATCH * INSZ * 2);
    _Float16* xdec    = (_Float16*)carve((size_t)BATCH * INSZ * 2);
    _Float16* hA      = (_Float16*)carve((size_t)BATCH * HID * 2);
    _Float16* hB      = (_Float16*)carve((size_t)BATCH * HID * 2);
    float*    cbuf    = (float*)carve((size_t)BATCH * HID * 4);

    const int T = 256;
    auto blocks = [&](int n) { return (n + T - 1) / T; };

    // ---- setup: precision conversion + bias fusion + state init ----
    cvt_f32_f16<<<blocks(4 * HID * HID), T, 0, stream>>>(enc_Whh, Whh_e16, 4 * HID * HID);
    cvt_f32_f16<<<blocks(4 * HID * HID), T, 0, stream>>>(dec_Whh, Whh_d16, 4 * HID * HID);
    cvt_f32_f16<<<blocks(4 * HID * INSZ), T, 0, stream>>>(enc_Wih, Wih_e16, 4 * HID * INSZ);
    cvt_f32_f16<<<blocks(4 * HID * INSZ), T, 0, stream>>>(dec_Wih, Wih_d16, 4 * HID * INSZ);
    cvt_f32_f16<<<blocks(INSZ * HID), T, 0, stream>>>(lin_W, linW16, INSZ * HID);
    cvt_f32_f16<<<blocks(SEQ * BATCH * INSZ), T, 0, stream>>>(input_batch, x16, SEQ * BATCH * INSZ);
    fuse_bias<<<blocks(4 * HID), T, 0, stream>>>(enc_bih, enc_bhh, bias_e, 4 * HID);
    fuse_bias<<<blocks(4 * HID), T, 0, stream>>>(dec_bih, dec_bhh, bias_d, 4 * HID);
    zero_state<<<blocks(BATCH * HID), T, 0, stream>>>(hA, cbuf, BATCH * HID);

    const dim3 stepGrid(16, 8);   // 64 col tiles / 4 waves, 8 row tiles

    // ---- encoder: 256 recurrent steps (h ping-pong, c in-place) ----
    _Float16* hp = hA;
    _Float16* hq = hB;
    for (int t = 0; t < SEQ; ++t) {
        lstm_step_kernel<<<stepGrid, 128, 0, stream>>>(
            x16 + (size_t)t * BATCH * INSZ,
            Wih_e16, Whh_e16, bias_e, hp, cbuf, hq);
        _Float16* tmp = hp; hp = hq; hq = tmp;
    }

    // ---- decoder: 96 recursive steps, x_{s+1} = y_s ----
    for (int s = 0; s < TGT; ++s) {
        const _Float16* xs = (s == 0) ? (x16 + (size_t)(SEQ - 1) * BATCH * INSZ) : xdec;
        lstm_step_kernel<<<stepGrid, 128, 0, stream>>>(
            xs, Wih_d16, Whh_d16, bias_d, hp, cbuf, hq);
        _Float16* tmp = hp; hp = hq; hq = tmp;
        dec_out_kernel<<<8, 128, 0, stream>>>(
            hp, linW16, lin_b, out + (size_t)s * BATCH * INSZ, xdec);
    }
}
